// CrAKNConvV1_69904887710284
// MI455X (gfx1250) — matrix operand order repaired and checked by reading, not compiled, not run on hardware
//
#include <hip/hip_runtime.h>
#include <hip/hip_bf16.h>

typedef __attribute__((ext_vector_type(16))) _Float16 v16h;
typedef __attribute__((ext_vector_type(8)))  float    v8f;

#define D 128
#define TROWS 64   // rows per workgroup tile (4 x 16-row WMMA stripes, one wave)

// ---------------------------------------------------------------------------
// Math helpers
// ---------------------------------------------------------------------------
__device__ __forceinline__ float mishf(float x) {
    float sp = (x > 20.0f) ? x : log1pf(__expf(x));   // softplus
    return x * tanhf(sp);
}

// ---------------------------------------------------------------------------
// WMMA fragment helpers (wave32, 16x16x32 f16 -> f32)
// A 16x32 f16 layout (ISA 7.12.2): m=lane&15; lanes<16: VGPR0..3 K=0..7,
// VGPR4..7 K=16..23 ; lanes>=16: K=8..15 and K=24..31.
// ---------------------------------------------------------------------------
__device__ __forceinline__ v16h load_a_frag(const _Float16* As, int kb) {
    const int lane = threadIdx.x & 31;
    const int m    = lane & 15;
    const int hi   = (lane >> 4) << 3;          // 0 or 8
    const _Float16* row = As + m * D + kb * 32;
    v16h a;
#pragma unroll
    for (int e = 0; e < 8; ++e) a[e]     = row[hi + e];
#pragma unroll
    for (int e = 0; e < 8; ++e) a[8 + e] = row[16 + hi + e];
    return a;
}

// B fragments come from a pre-packed weight layout: one contiguous v16h per lane.
__device__ __forceinline__ v16h load_b_frag(const _Float16* Wp, int kb, int nb) {
    const int lane = threadIdx.x & 31;
    return *(const v16h*)(Wp + ((((kb * 8 + nb) * 32) + lane) << 4));
}

// C/D 16x16 f32 layout: VGPR v -> M = v + (lane>=16 ? 8 : 0), N = lane&15.
__device__ __forceinline__ void store_c(float* Cs, int nb, v8f c) {
    const int lane = threadIdx.x & 31;
    const int n    = nb * 16 + (lane & 15);
    const int mb   = (lane >> 4) << 3;
#pragma unroll
    for (int vi = 0; vi < 8; ++vi) Cs[(mb + vi) * D + n] = c[vi];
}

__device__ __forceinline__ v8f load_c(const float* Cs, int nb) {
    const int lane = threadIdx.x & 31;
    const int n    = nb * 16 + (lane & 15);
    const int mb   = (lane >> 4) << 3;
    v8f c;
#pragma unroll
    for (int vi = 0; vi < 8; ++vi) c[vi] = Cs[(mb + vi) * D + n];
    return c;
}

// One wave computes a 64x128 = [64,128] @ [128,128] tile.
// Each B fragment is reused across 4 row-stripes: 16 wmma per 8 b128 loads.
template <bool ACC>
__device__ __forceinline__ void gemm_64x128x128(const _Float16* As,
                                                const _Float16* Wp,
                                                float* Cs) {
    v16h a[4][4];
#pragma unroll
    for (int s = 0; s < 4; ++s)
#pragma unroll
        for (int kb = 0; kb < 4; ++kb)
            a[s][kb] = load_a_frag(As + s * 16 * D, kb);

#pragma unroll
    for (int nb = 0; nb < 8; ++nb) {
        v16h b0 = load_b_frag(Wp, 0, nb);
        v16h b1 = load_b_frag(Wp, 1, nb);
        v16h b2 = load_b_frag(Wp, 2, nb);
        v16h b3 = load_b_frag(Wp, 3, nb);
#pragma unroll
        for (int s = 0; s < 4; ++s) {
            v8f c = {};
            if (ACC) c = load_c(Cs + s * 16 * D, nb);
            c = __builtin_amdgcn_wmma_f32_16x16x32_f16(false, a[s][0], false, b0, (short)0, c, false, false);
            c = __builtin_amdgcn_wmma_f32_16x16x32_f16(false, a[s][1], false, b1, (short)0, c, false, false);
            c = __builtin_amdgcn_wmma_f32_16x16x32_f16(false, a[s][2], false, b2, (short)0, c, false, false);
            c = __builtin_amdgcn_wmma_f32_16x16x32_f16(false, a[s][3], false, b3, (short)0, c, false, false);
            store_c(Cs + s * 16 * D, nb, c);
        }
    }
}

// Prefetch a packed 128x128 weight block (32KB) into cache (global_prefetch_b8).
__device__ __forceinline__ void prefetch_weights(const _Float16* Wp) {
    const int lane = threadIdx.x & 31;
#pragma unroll
    for (int i = 0; i < 4; ++i)
        __builtin_prefetch(Wp + (size_t)(i * 32 + lane) * 128, 0, 0);
}

// ---------------------------------------------------------------------------
// Epilogues: 32 lanes over a 64x128 LDS tile; each lane owns 2 full rows,
// so LayerNorm stats are lane-local (no cross-lane reduction needed).
// ---------------------------------------------------------------------------

// bias -> LayerNorm -> Mish -> f16 LDS (feeds next GEMM)
__device__ __forceinline__ void epi_ln_mish_f16(const float* Cs, const float* b,
                                                const float* g, const float* be,
                                                _Float16* out) {
#pragma unroll
    for (int rr = 0; rr < 2; ++rr) {
        const int r = threadIdx.x * 2 + rr;
        float s = 0.f, s2 = 0.f;
#pragma unroll 4
        for (int c = 0; c < D; ++c) {
            float v = Cs[r * D + c] + b[c];
            s += v; s2 += v * v;
        }
        float mean = s * (1.0f / D);
        float var  = s2 * (1.0f / D) - mean * mean;
        float rstd = rsqrtf(var + 1e-5f);
#pragma unroll 4
        for (int c = 0; c < D; ++c) {
            float v = Cs[r * D + c] + b[c];
            v = (v - mean) * rstd * g[c] + be[c];
            out[r * D + c] = (_Float16)mishf(v);
        }
    }
}

// bias -> LayerNorm -> Mish -> f32 global
__device__ __forceinline__ void epi_ln_mish_f32g(const float* Cs, const float* b,
                                                 const float* g, const float* be,
                                                 float* out, int rowbase, int nrows) {
#pragma unroll
    for (int rr = 0; rr < 2; ++rr) {
        const int r = threadIdx.x * 2 + rr;
        if (rowbase + r >= nrows) continue;
        float s = 0.f, s2 = 0.f;
#pragma unroll 4
        for (int c = 0; c < D; ++c) {
            float v = Cs[r * D + c] + b[c];
            s += v; s2 += v * v;
        }
        float mean = s * (1.0f / D);
        float var  = s2 * (1.0f / D) - mean * mean;
        float rstd = rsqrtf(var + 1e-5f);
#pragma unroll 4
        for (int c = 0; c < D; ++c) {
            float v = Cs[r * D + c] + b[c];
            v = (v - mean) * rstd * g[c] + be[c];
            out[(size_t)(rowbase + r) * D + c] = mishf(v);
        }
    }
}

// bias -> Mish -> f16 LDS
__device__ __forceinline__ void epi_mish_f16(const float* Cs, const float* b,
                                             _Float16* out) {
#pragma unroll
    for (int rr = 0; rr < 2; ++rr) {
        const int r = threadIdx.x * 2 + rr;
#pragma unroll 4
        for (int c = 0; c < D; ++c)
            out[r * D + c] = (_Float16)mishf(Cs[r * D + c] + b[c]);
    }
}

// (+bias) -> f32 global
__device__ __forceinline__ void epi_store_f32g(const float* Cs, const float* b,
                                               float* out, int rowbase, int nrows) {
#pragma unroll
    for (int rr = 0; rr < 2; ++rr) {
        const int r = threadIdx.x * 2 + rr;
        if (rowbase + r >= nrows) continue;
#pragma unroll 4
        for (int c = 0; c < D; ++c) {
            float v = Cs[r * D + c];
            if (b) v += b[c];
            out[(size_t)(rowbase + r) * D + c] = v;
        }
    }
}

// Coalesced global f32 -> LDS f16 staging of a 64x128 tile (float4 loads).
__device__ __forceinline__ void stage_f16(const float* src, _Float16* dstLds,
                                          int rowbase, int nrows) {
    const float4* s4 = (const float4*)src;
#pragma unroll 4
    for (int j = 0; j < 64; ++j) {
        int f  = threadIdx.x + 32 * j;       // 0..2047 float4 slots
        int r  = f >> 5;                     // tile row
        int q4 = f & 31;                     // float4 within row
        int row = rowbase + r; if (row >= nrows) row = nrows - 1;
        float4 v = s4[(size_t)row * 32 + q4];
        _Float16* p = dstLds + r * D + q4 * 4;
        p[0] = (_Float16)v.x; p[1] = (_Float16)v.y;
        p[2] = (_Float16)v.z; p[3] = (_Float16)v.w;
    }
}

// ---------------------------------------------------------------------------
// Kernel: pack a 128x128 f32 weight block into the B-fragment-native f16 layout
// packed[((kb*8+nb)*32+lane)*16+e] = W[kb*32 + (lane>=16?8:0) + (e&7) + (e>=8?16:0)][nb*16 + (lane&15)]
// ---------------------------------------------------------------------------
__global__ void pack_weight_128(const float* __restrict__ W, _Float16* __restrict__ Wp) {
    int p = blockIdx.x * blockDim.x + threadIdx.x;   // 0..16383
    int e    = p & 15;
    int lane = (p >> 4) & 31;
    int nb   = (p >> 9) & 7;
    int kb   = p >> 12;
    int row  = kb * 32 + ((lane >> 4) << 3) + (e & 7) + ((e >> 3) << 4);
    int col  = nb * 16 + (lane & 15);
    Wp[p] = (_Float16)W[row * D + col];
}

// ---------------------------------------------------------------------------
// Node stage 1: q, k, pem, peb, x  (all from coord_feat); 64 nodes per wave.
// ---------------------------------------------------------------------------
__global__ void __launch_bounds__(32)
node_stage1(const float* __restrict__ coord,
            const float* bq, const float* gq, const float* beq,
            const float* bk, const float* gk, const float* bek,
            const float* bm1, const float* gm, const float* bem, const float* bm2,
            const float* bb1, const float* gb, const float* beb, const float* bb2,
            const float* bc1,
            const _Float16* Wqp, const _Float16* Wkp,
            const _Float16* Wm1p, const _Float16* Wm2p,
            const _Float16* Wb1p, const _Float16* Wb2p,
            const _Float16* Wc1p, const _Float16* Wc2p,
            float* qout, float* kout, float* pemout, float* pebout, float* xout,
            int N) {
    __shared__ _Float16 As[TROWS * D];
    __shared__ _Float16 Bs[TROWS * D];
    __shared__ float    Cs[TROWS * D];

    const int rowbase = blockIdx.x * TROWS;
    stage_f16(coord, As, rowbase, N);
    prefetch_weights(Wqp);
    __syncthreads();

    // q
    gemm_64x128x128<false>(As, Wqp, Cs); __syncthreads();
    prefetch_weights(Wkp);
    epi_ln_mish_f32g(Cs, bq, gq, beq, qout, rowbase, N); __syncthreads();
    // k
    gemm_64x128x128<false>(As, Wkp, Cs); __syncthreads();
    prefetch_weights(Wm1p);
    epi_ln_mish_f32g(Cs, bk, gk, bek, kout, rowbase, N); __syncthreads();
    // pem = mish(ln(coord@Wm1+bm1)) @ Wm2 + bm2
    gemm_64x128x128<false>(As, Wm1p, Cs); __syncthreads();
    prefetch_weights(Wm2p);
    epi_ln_mish_f16(Cs, bm1, gm, bem, Bs); __syncthreads();
    gemm_64x128x128<false>(Bs, Wm2p, Cs); __syncthreads();
    prefetch_weights(Wb1p);
    epi_store_f32g(Cs, bm2, pemout, rowbase, N); __syncthreads();
    // peb
    gemm_64x128x128<false>(As, Wb1p, Cs); __syncthreads();
    prefetch_weights(Wb2p);
    epi_ln_mish_f16(Cs, bb1, gb, beb, Bs); __syncthreads();
    gemm_64x128x128<false>(Bs, Wb2p, Cs); __syncthreads();
    prefetch_weights(Wc1p);
    epi_store_f32g(Cs, bb2, pebout, rowbase, N); __syncthreads();
    // x = mish(coord@Wc1+bc1) @ Wc2
    gemm_64x128x128<false>(As, Wc1p, Cs); __syncthreads();
    prefetch_weights(Wc2p);
    epi_mish_f16(Cs, bc1, Bs); __syncthreads();
    gemm_64x128x128<false>(Bs, Wc2p, Cs); __syncthreads();
    epi_store_f32g(Cs, nullptr, xout, rowbase, N);
}

// ---------------------------------------------------------------------------
// Edge kernel: rel -> normalize -> *pem+peb -> mish(ln(rel@Ww1+bw1))@Ww2+bw2
// 64 edges per wave; writes raw logits to wbuf [E,128].
// ---------------------------------------------------------------------------
__global__ void __launch_bounds__(32)
edge_kernel(const float* __restrict__ q, const float* __restrict__ kf,
            const float* __restrict__ pem, const float* __restrict__ peb,
            const float* bw1, const float* gw, const float* bew, const float* bw2,
            const _Float16* Ww1p, const _Float16* Ww2p,
            const int* __restrict__ src, const int* __restrict__ dst,
            float* __restrict__ wout, int E) {
    __shared__ _Float16 As[TROWS * D];
    __shared__ _Float16 Bs[TROWS * D];
    __shared__ float    Cs[TROWS * D];

    const int ebase = blockIdx.x * TROWS;
    prefetch_weights(Ww1p);

#pragma unroll
    for (int rr = 0; rr < 2; ++rr) {
        const int r = threadIdx.x * 2 + rr;
        int erow = ebase + r; if (erow >= E) erow = E - 1;
        const int s = src[erow], d = dst[erow];
        float ssq = 0.f;
#pragma unroll 4
        for (int c = 0; c < D; ++c) {
            float v = q[(size_t)s * D + c] - kf[(size_t)d * D + c];
            Cs[r * D + c] = v;
            ssq += v * v;
        }
        float rn = rsqrtf(ssq + 1e-8f);
#pragma unroll 4
        for (int c = 0; c < D; ++c) {
            float v = Cs[r * D + c] * rn;
            v = v * pem[(size_t)d * D + c] + peb[(size_t)d * D + c];
            As[r * D + c] = (_Float16)v;
        }
    }
    __syncthreads();

    gemm_64x128x128<false>(As, Ww1p, Cs); __syncthreads();
    prefetch_weights(Ww2p);
    epi_ln_mish_f16(Cs, bw1, gw, bew, Bs); __syncthreads();
    gemm_64x128x128<false>(Bs, Ww2p, Cs); __syncthreads();
    epi_store_f32g(Cs, bw2, wout, ebase, E);
}

// ---------------------------------------------------------------------------
// Column softmax over E edges (axis=0), 128 columns
// ---------------------------------------------------------------------------
__global__ void colmax_partial(const float* __restrict__ w, float* __restrict__ part,
                               long long E) {
    const int c = threadIdx.x;
    float m = -3.402823466e38f;
    for (long long r = blockIdx.x; r < E; r += gridDim.x)
        m = fmaxf(m, w[r * D + c]);
    part[(size_t)blockIdx.x * D + c] = m;
}

__global__ void colmax_reduce(const float* __restrict__ part, float* __restrict__ colmax,
                              int npart) {
    const int c = threadIdx.x;
    float m = -3.402823466e38f;
    for (int i = 0; i < npart; ++i) m = fmaxf(m, part[(size_t)i * D + c]);
    colmax[c] = m;
}

__global__ void exp_colsum(float* __restrict__ w, const float* __restrict__ colmax,
                           float* __restrict__ colsum, long long E) {
    const int c = threadIdx.x;
    const float mx = colmax[c];
    float acc = 0.f;
    for (long long r = blockIdx.x; r < E; r += gridDim.x) {
        float v = __expf(w[r * D + c] - mx);
        w[r * D + c] = v;
        acc += v;
    }
    atomicAdd(&colsum[c], acc);
}

__global__ void inv_colsum(float* colsum) {
    const int c = threadIdx.x;
    colsum[c] = 1.0f / colsum[c];
}

__global__ void scatter_edges(const float* __restrict__ w, const float* __restrict__ inv,
                              const int* __restrict__ dst, float* __restrict__ hn,
                              long long total) {
    long long idx = (long long)blockIdx.x * blockDim.x + threadIdx.x;
    const long long stride = (long long)gridDim.x * blockDim.x;
    for (; idx < total; idx += stride) {
        int c = (int)(idx & (D - 1));
        long long r = idx >> 7;
        atomicAdd(&hn[(size_t)dst[r] * D + c], w[idx] * inv[c]);
    }
}

// ---------------------------------------------------------------------------
// Node stage 2: h = mish(concat(node_feat, h_neigh) @ Wn1 + bn1) @ Wn2 + bn2
// ---------------------------------------------------------------------------
__global__ void __launch_bounds__(32)
node_stage2(const float* __restrict__ node_feat, const float* __restrict__ hn,
            const float* bn1, const float* bn2,
            const _Float16* Wn1ap, const _Float16* Wn1bp, const _Float16* Wn2p,
            float* __restrict__ hout, int N) {
    __shared__ _Float16 As[TROWS * D];
    __shared__ _Float16 Bs[TROWS * D];
    __shared__ float    Cs[TROWS * D];

    const int rowbase = blockIdx.x * TROWS;
    prefetch_weights(Wn1ap);
    stage_f16(node_feat, As, rowbase, N);
    stage_f16(hn,        Bs, rowbase, N);
    __syncthreads();

    gemm_64x128x128<false>(As, Wn1ap, Cs); __syncthreads();
    prefetch_weights(Wn1bp);
    gemm_64x128x128<true >(Bs, Wn1bp, Cs); __syncthreads();
    prefetch_weights(Wn2p);
    epi_mish_f16(Cs, bn1, As); __syncthreads();
    gemm_64x128x128<false>(As, Wn2p, Cs); __syncthreads();
    epi_store_f32g(Cs, bn2, hout, rowbase, N);
}

// ---------------------------------------------------------------------------
// Launch
// ---------------------------------------------------------------------------
extern "C" void kernel_launch(void* const* d_in, const int* in_sizes, int n_in,
                              void* d_out, int out_size, void* d_ws, size_t ws_size,
                              hipStream_t stream) {
    const float* node_feat = (const float*)d_in[0];
    const float* coord     = (const float*)d_in[1];
    const float* Wq  = (const float*)d_in[2];  const float* bq  = (const float*)d_in[3];
    const float* gq  = (const float*)d_in[4];  const float* beq = (const float*)d_in[5];
    const float* Wk  = (const float*)d_in[6];  const float* bk  = (const float*)d_in[7];
    const float* gk  = (const float*)d_in[8];  const float* bek = (const float*)d_in[9];
    const float* Wm1 = (const float*)d_in[10]; const float* bm1 = (const float*)d_in[11];
    const float* gm  = (const float*)d_in[12]; const float* bem = (const float*)d_in[13];
    const float* Wm2 = (const float*)d_in[14]; const float* bm2 = (const float*)d_in[15];
    const float* Wb1 = (const float*)d_in[16]; const float* bb1 = (const float*)d_in[17];
    const float* gb  = (const float*)d_in[18]; const float* beb = (const float*)d_in[19];
    const float* Wb2 = (const float*)d_in[20]; const float* bb2 = (const float*)d_in[21];
    const float* Ww1 = (const float*)d_in[22]; const float* bw1 = (const float*)d_in[23];
    const float* gw  = (const float*)d_in[24]; const float* bew = (const float*)d_in[25];
    const float* Ww2 = (const float*)d_in[26]; const float* bw2 = (const float*)d_in[27];
    const float* Wn1 = (const float*)d_in[28]; const float* bn1 = (const float*)d_in[29];
    const float* Wn2 = (const float*)d_in[30]; const float* bn2 = (const float*)d_in[31];
    const float* Wc1 = (const float*)d_in[32]; const float* bc1 = (const float*)d_in[33];
    const float* Wc2 = (const float*)d_in[34];
    const int*   src = (const int*)d_in[35];
    const int*   dst = (const int*)d_in[36];

    const int N = in_sizes[0] / D;
    const int E = in_sizes[35];

    // ---- workspace carve-up (256B aligned) ----
    size_t off = 0;
    auto carve = [&](size_t bytes) {
        size_t cur = off;
        off += (bytes + 255) & ~(size_t)255;
        return (char*)d_ws + cur;
    };
    const size_t WBLK = 16384;                      // elements per packed 128x128 block
    _Float16* PW    = (_Float16*)carve(13 * WBLK * sizeof(_Float16));
    float* qbuf     = (float*)carve((size_t)N * D * 4);
    float* kbuf     = (float*)carve((size_t)N * D * 4);
    float* pembuf   = (float*)carve((size_t)N * D * 4);
    float* pebbuf   = (float*)carve((size_t)N * D * 4);
    float* hneigh   = (float*)carve((size_t)N * D * 4);
    float* wbuf     = (float*)carve((size_t)E * D * 4);
    float* partbuf  = (float*)carve((size_t)256 * D * 4);
    float* colmax   = (float*)carve(D * 4);
    float* colsum   = (float*)carve(D * 4);

    // ---- 1. pack weights to f16 fragment layout ----
    const float* wsrc[13] = {Wq, Wk, Wm1, Wm2, Wb1, Wb2, Ww1, Ww2,
                             Wn1, Wn1 + (size_t)D * D, Wn2, Wc1, Wc2};
    for (int i = 0; i < 13; ++i)
        pack_weight_128<<<WBLK / 256, 256, 0, stream>>>(wsrc[i], PW + i * WBLK);
    _Float16* Wqp  = PW + 0 * WBLK; _Float16* Wkp  = PW + 1 * WBLK;
    _Float16* Wm1p = PW + 2 * WBLK; _Float16* Wm2p = PW + 3 * WBLK;
    _Float16* Wb1p = PW + 4 * WBLK; _Float16* Wb2p = PW + 5 * WBLK;
    _Float16* Ww1p = PW + 6 * WBLK; _Float16* Ww2p = PW + 7 * WBLK;
    _Float16* Wn1ap = PW + 8 * WBLK; _Float16* Wn1bp = PW + 9 * WBLK;
    _Float16* Wn2p = PW + 10 * WBLK;
    _Float16* Wc1p = PW + 11 * WBLK; _Float16* Wc2p = PW + 12 * WBLK;

    // ---- zero accumulators ----
    hipMemsetAsync(hneigh, 0, (size_t)N * D * 4, stream);
    hipMemsetAsync(colsum, 0, D * 4, stream);

    float* hout = (float*)d_out;                    // h: [N,128]
    float* xout = (float*)d_out + (size_t)N * D;    // x: [N,128]

    // ---- 2. node stage 1 (q,k,pem,peb,x) ----
    int nodeBlocks = (N + TROWS - 1) / TROWS;
    node_stage1<<<nodeBlocks, 32, 0, stream>>>(
        coord, bq, gq, beq, bk, gk, bek, bm1, gm, bem, bm2,
        bb1, gb, beb, bb2, bc1,
        Wqp, Wkp, Wm1p, Wm2p, Wb1p, Wb2p, Wc1p, Wc2p,
        qbuf, kbuf, pembuf, pebbuf, xout, N);

    // ---- 3. edge MLP -> raw logits ----
    int edgeBlocks = (E + TROWS - 1) / TROWS;
    edge_kernel<<<edgeBlocks, 32, 0, stream>>>(
        qbuf, kbuf, pembuf, pebbuf, bw1, gw, bew, bw2,
        Ww1p, Ww2p, src, dst, wbuf, E);

    // ---- 4. column softmax ----
    colmax_partial<<<256, D, 0, stream>>>(wbuf, partbuf, (long long)E);
    colmax_reduce<<<1, D, 0, stream>>>(partbuf, colmax, 256);
    exp_colsum<<<256, D, 0, stream>>>(wbuf, colmax, colsum, (long long)E);
    inv_colsum<<<1, D, 0, stream>>>(colsum);

    // ---- 5. scatter-add to destination nodes ----
    scatter_edges<<<2048, 256, 0, stream>>>(wbuf, colsum, dst, hneigh,
                                            (long long)E * D);

    // ---- 6. node stage 2 (h) ----
    node_stage2<<<nodeBlocks, 32, 0, stream>>>(
        node_feat, hneigh, bn1, bn2, Wn1ap, Wn1bp, Wn2p, hout, N);
}
